// SelfAttentionBlock_4011499455113
// MI455X (gfx1250) — compile-verified, matrix operand order checked by reading, not run on hardware
//
#include <hip/hip_runtime.h>

typedef __attribute__((ext_vector_type(8)))  __bf16 v8bf;
typedef __attribute__((ext_vector_type(16))) __bf16 v16bf;
typedef __attribute__((ext_vector_type(8)))  float  v8f;
typedef int v4i __attribute__((vector_size(4 * sizeof(int))));

#define GLOBAL_AS __attribute__((address_space(1)))
#define LDS_AS    __attribute__((address_space(3)))

#if defined(__gfx1250__) &&                                        \
    __has_builtin(__builtin_amdgcn_global_load_async_to_lds_b128) && \
    __has_builtin(__builtin_amdgcn_s_wait_asynccnt)
#define HAVE_ASYNC_LDS 1
#else
#define HAVE_ASYNC_LDS 0
#endif

// ---------------------------------------------------------------------------
// Operand loader for 16-bit WMMA A/B tiles (16x32 A, or 32x16 B stored
// column-major as [N][K]). Lane (m = lane&15, half = lane>>4) holds 16 bf16:
//   elems[0..7]  = K 8*half   .. 8*half+7
//   elems[8..15] = K 16+8*half .. 16+8*half+7
// Both chunks are 16B-aligned contiguous loads -> b128.
// ---------------------------------------------------------------------------
__device__ __forceinline__ v16bf load_tile_bf16(const __bf16* rowptr, int half) {
  const v8bf a = *(const v8bf*)(rowptr + half * 8);
  const v8bf b = *(const v8bf*)(rowptr + 16 + half * 8);
  v16bf r;
#pragma unroll
  for (int i = 0; i < 8; ++i) { r[i] = a[i]; r[8 + i] = b[i]; }
  return r;
}

// Same tile pattern but source is f32 (LDS softmax output) converted to bf16.
__device__ __forceinline__ v16bf load_tile_f32cvt(const float* rowptr, int half) {
  v16bf r;
#pragma unroll
  for (int i = 0; i < 8; ++i) {
    r[i]     = (__bf16)rowptr[8 * half + i];
    r[8 + i] = (__bf16)rowptr[16 + 8 * half + i];
  }
  return r;
}

__device__ __forceinline__ v8f wmma_bf16(v16bf a, v16bf b, v8f c) {
  return __builtin_amdgcn_wmma_f32_16x16x32_bf16(false, a, false, b, (short)0, c,
                                                 false, false);
}

// ---------------------------------------------------------------------------
// Kernel 0: pack weights to bf16.
//   wcat[192][256]: rows 0..31 w_theta, 32..63 w_phi, 64..191 w_g  (A operand)
//   wo_b[256][128]: w_o as-is (B operand: lane = cout row, K = c2 contiguous)
// ---------------------------------------------------------------------------
__global__ void k_pack_weights(const float* __restrict__ w_theta,
                               const float* __restrict__ w_phi,
                               const float* __restrict__ w_g,
                               const float* __restrict__ w_o,
                               __bf16* __restrict__ wcat,
                               __bf16* __restrict__ wo_b) {
  int i = blockIdx.x * 256 + threadIdx.x;
  if (i < 192 * 256) {
    int r = i >> 8, c = i & 255;
    float v = (r < 32)   ? w_theta[r * 256 + c]
              : (r < 64) ? w_phi[(r - 32) * 256 + c]
                         : w_g[(r - 64) * 256 + c];
    wcat[i] = (__bf16)v;
  } else if (i < 192 * 256 + 256 * 128) {
    int j = i - 192 * 256;
    wo_b[j] = (__bf16)w_o[j];
  }
}

// ---------------------------------------------------------------------------
// Kernel 1: projection GEMM  P[192,4096] = Wcat[192,256] x X[b][256,4096]
// One WG per (b, 64-pixel block). x tile staged to LDS transposed as
// bf16 [64 px][256 c]. 8 waves: wave w -> n-tile (w&3), m-tiles (w>>2)*6+0..5.
// Outputs routed by channel:
//   thetaT [b][4096][32]  (scores A operand rows = t)
//   phi_pre[b][32][4096], g_pre[b][128][4096]  (pooled later)
// ---------------------------------------------------------------------------
__global__ void k_proj(const float* __restrict__ x,
                       const __bf16* __restrict__ wcat,
                       __bf16* __restrict__ thetaT,
                       __bf16* __restrict__ phi_pre,
                       __bf16* __restrict__ g_pre) {
  extern __shared__ char smem[];
  __bf16* xt = (__bf16*)smem;  // [64][256] bf16 = 32KB

  const int b   = blockIdx.y;
  const int pxb = blockIdx.x * 64;
  const int tid = threadIdx.x;

  // Stage + convert: each thread owns one px, 64 consecutive channels.
  {
    int px = tid & 63;
    int cb = (tid >> 6) * 64;
    const float* xb = x + ((size_t)b * 256) * 4096 + pxb + px;
#pragma unroll 8
    for (int i = 0; i < 64; ++i) {
      int c = cb + i;
      if ((i & 7) == 0) __builtin_prefetch(xb + (size_t)c * 4096 + 64, 0, 0);
      xt[px * 256 + c] = (__bf16)xb[(size_t)c * 4096];
    }
  }
  __syncthreads();

  const int w = tid >> 5, lane = tid & 31;
  const int lanelow = lane & 15, half = lane >> 4;
  const int nt = w & 3;
  const int mg = (w >> 2) * 6;

  v8f acc[6] = {};
  for (int kc = 0; kc < 256; kc += 32) {
    v16bf bmat = load_tile_bf16(xt + (nt * 16 + lanelow) * 256 + kc, half);
#pragma unroll
    for (int j = 0; j < 6; ++j) {
      v16bf amat =
          load_tile_bf16(wcat + ((mg + j) * 16 + lanelow) * 256 + kc, half);
      acc[j] = wmma_bf16(amat, bmat, acc[j]);
    }
  }

  const int pxg = pxb + nt * 16 + lanelow;  // D col n
#pragma unroll
  for (int j = 0; j < 6; ++j) {
    int mt = mg + j;
#pragma unroll
    for (int r = 0; r < 8; ++r) {
      int cout = mt * 16 + r + half * 8;  // D row m
      __bf16 v = (__bf16)acc[j][r];
      if (cout < 32)
        thetaT[((size_t)b * 4096 + pxg) * 32 + cout] = v;
      else if (cout < 64)
        phi_pre[((size_t)b * 32 + (cout - 32)) * 4096 + pxg] = v;
      else
        g_pre[((size_t)b * 128 + (cout - 64)) * 4096 + pxg] = v;
    }
  }
}

// ---------------------------------------------------------------------------
// Kernel 2: 2x2 maxpool.
//   phi_p[b][1024][32] (B operand for scores: lane = p, K = c contiguous)
//   g_p  [b][128][1024] (B operand for attn: lane = c2, K = p contiguous)
// ---------------------------------------------------------------------------
__global__ void k_pool(const __bf16* __restrict__ phi_pre,
                       const __bf16* __restrict__ g_pre,
                       __bf16* __restrict__ phi_p,
                       __bf16* __restrict__ g_p) {
  int i = blockIdx.x * 256 + threadIdx.x;
  const int NPHI = 16 * 1024 * 32;
  if (i < NPHI) {
    int b = i >> 15, rem = i & 32767;
    int p = rem >> 5, c = rem & 31;
    int ph = p >> 5, pw = p & 31;
    size_t base = ((size_t)b * 32 + c) * 4096 + (size_t)(ph * 2) * 64 + pw * 2;
    float v = fmaxf(fmaxf((float)phi_pre[base], (float)phi_pre[base + 1]),
                    fmaxf((float)phi_pre[base + 64], (float)phi_pre[base + 65]));
    phi_p[((size_t)b * 1024 + p) * 32 + c] = (__bf16)v;
  } else {
    int j = i - NPHI;
    if (j >= 16 * 128 * 1024) return;
    int b = j >> 17, rem = j & 131071;
    int c2 = rem >> 10, p = rem & 1023;
    int ph = p >> 5, pw = p & 31;
    size_t base = ((size_t)b * 128 + c2) * 4096 + (size_t)(ph * 2) * 64 + pw * 2;
    float v = fmaxf(fmaxf((float)g_pre[base], (float)g_pre[base + 1]),
                    fmaxf((float)g_pre[base + 64], (float)g_pre[base + 65]));
    g_p[((size_t)b * 128 + c2) * 1024 + p] = (__bf16)v;
  }
}

// ---------------------------------------------------------------------------
// Kernel 3: fused attention. One WG = (batch b, 32-row t-block).
// LDS layout (bytes):
//   [0,128K)        scores[32][1024] f32
//   [128K,136K)     attn[32][128] bf16
//   [136K,137K)     red[256] f32
//   [137K,201K)     phi stage [1024][32] bf16 (async global->LDS copy)
// Phase A: scores via single-K wmma (C8==32), phi fed from LDS.
// Phase B: softmax in LDS.  Phase C: attn = beta x g.  Phase D: w_o GEMM +
// gamma*o + x residual.
// ---------------------------------------------------------------------------
#define SC_BYTES   (32 * 1024 * 4)
#define AT_BYTES   (32 * 128 * 2)
#define RED_BYTES  (256 * 4)
#define PHI_BYTES  (1024 * 32 * 2)
#define ATTN_LDS   (SC_BYTES + AT_BYTES + RED_BYTES + PHI_BYTES)

__global__ void k_attn(const float* __restrict__ x,
                       const __bf16* __restrict__ thetaT,
                       const __bf16* __restrict__ phi_p,
                       const __bf16* __restrict__ g_p,
                       const __bf16* __restrict__ wo_b,
                       const float* __restrict__ gamma_p,
                       float* __restrict__ out) {
  extern __shared__ char smem[];
  float*  scores = (float*)smem;
  __bf16* attn_l = (__bf16*)(smem + SC_BYTES);
  float*  red    = (float*)(smem + SC_BYTES + AT_BYTES);
  __bf16* phi_l  = (__bf16*)(smem + SC_BYTES + AT_BYTES + RED_BYTES);

  const int b    = blockIdx.y;
  const int tblk = blockIdx.x * 32;
  const int tid  = threadIdx.x;
  const int w = tid >> 5, lane = tid & 31;
  const int lanelow = lane & 15, half = lane >> 4;

  // ---- Stage phi[b] (64KB) into LDS via CDNA5 async global->LDS copy ----
  {
    const char* gsrc = (const char*)(phi_p + (size_t)b * 1024 * 32);
    char* ldst = (char*)phi_l;
#if HAVE_ASYNC_LDS
#pragma unroll
    for (int i = 0; i < 16; ++i) {
      int off = tid * 16 + i * 4096;  // 256 threads x 16B x 16 iters = 64KB
      __builtin_amdgcn_global_load_async_to_lds_b128(
          (GLOBAL_AS v4i*)(gsrc + off), (LDS_AS v4i*)(ldst + off), 0, 0);
    }
    __builtin_amdgcn_s_wait_asynccnt(0);
#else
#pragma unroll
    for (int i = 0; i < 16; ++i) {
      int off = tid * 16 + i * 4096;
      *(v8bf*)(ldst + off) = *(const v8bf*)(gsrc + off);
    }
#endif
  }
  __syncthreads();

  // ---- Phase A: scores[32][1024] ----
  {
    const int tt = w & 1;
    v16bf amat = load_tile_bf16(
        thetaT + ((size_t)b * 4096 + tblk + tt * 16 + lanelow) * 32, half);
    const int pt0 = (w >> 1) * 16;
#pragma unroll 4
    for (int i = 0; i < 16; ++i) {
      int pt = pt0 + i;
      v16bf bmat = load_tile_bf16(phi_l + (pt * 16 + lanelow) * 32, half);
      v8f z = {};
      v8f d = wmma_bf16(amat, bmat, z);
#pragma unroll
      for (int r = 0; r < 8; ++r)
        scores[(tt * 16 + r + half * 8) * 1024 + pt * 16 + lanelow] = d[r];
    }
  }
  __syncthreads();

  // ---- Phase B: row softmax (8 threads per row, 128 cols each) ----
  {
    const int row = tid >> 3, sub = tid & 7;
    float* srow = scores + row * 1024 + sub * 128;
    float m = -3.4e38f;
    for (int c = 0; c < 128; ++c) m = fmaxf(m, srow[c]);
    red[tid] = m;
    __syncthreads();
    float rm = red[row * 8];
#pragma unroll
    for (int k = 1; k < 8; ++k) rm = fmaxf(rm, red[row * 8 + k]);
    __syncthreads();
    float s = 0.f;
    for (int c = 0; c < 128; ++c) {
      float e = __expf(srow[c] - rm);
      srow[c] = e;
      s += e;
    }
    red[tid] = s;
    __syncthreads();
    float tot = 0.f;
#pragma unroll
    for (int k = 0; k < 8; ++k) tot += red[row * 8 + k];
    float inv = 1.0f / tot;
    for (int c = 0; c < 128; ++c) srow[c] *= inv;
  }
  __syncthreads();

  // ---- Phase C: attn[32][128] = beta x g^T ----
  {
    const int tt = w & 1;
    const int ctb = (w >> 1) * 2;  // two c2 tiles per wave
    v8f acc0 = {}, acc1 = {};
    for (int kc = 0; kc < 1024; kc += 32) {
      v16bf amat =
          load_tile_f32cvt(scores + (tt * 16 + lanelow) * 1024 + kc, half);
      v16bf b0 = load_tile_bf16(
          g_p + ((size_t)b * 128 + ctb * 16 + lanelow) * 1024 + kc, half);
      v16bf b1 = load_tile_bf16(
          g_p + ((size_t)b * 128 + (ctb + 1) * 16 + lanelow) * 1024 + kc, half);
      acc0 = wmma_bf16(amat, b0, acc0);
      acc1 = wmma_bf16(amat, b1, acc1);
    }
#pragma unroll
    for (int r = 0; r < 8; ++r) {
      int t_loc = tt * 16 + r + half * 8;
      attn_l[t_loc * 128 + ctb * 16 + lanelow]       = (__bf16)acc0[r];
      attn_l[t_loc * 128 + (ctb + 1) * 16 + lanelow] = (__bf16)acc1[r];
    }
  }
  __syncthreads();

  // ---- Phase D: o = w_o x attn ; out = gamma*o + x ----
  {
    const int tt = w & 1;
    const int obase = (w >> 1) * 4;  // four cout tiles per wave
    v8f acc[4] = {};
#pragma unroll
    for (int kc = 0; kc < 128; kc += 32) {
      v16bf amat = load_tile_bf16(attn_l + (tt * 16 + lanelow) * 128 + kc, half);
#pragma unroll
      for (int q = 0; q < 4; ++q) {
        v16bf bmat = load_tile_bf16(
            wo_b + ((obase + q) * 16 + lanelow) * 128 + kc, half);
        acc[q] = wmma_bf16(amat, bmat, acc[q]);
      }
    }
    const float gm = gamma_p[0];
#pragma unroll
    for (int q = 0; q < 4; ++q) {
      int cout = (obase + q) * 16 + lanelow;  // D col n
#pragma unroll
      for (int r = 0; r < 8; ++r) {
        int t = tblk + tt * 16 + r + half * 8;  // D row m
        size_t idx = ((size_t)b * 256 + cout) * 4096 + t;
        out[idx] = gm * acc[q][r] + x[idx];
      }
    }
  }
}

// ---------------------------------------------------------------------------
extern "C" void kernel_launch(void* const* d_in, const int* in_sizes, int n_in,
                              void* d_out, int out_size, void* d_ws,
                              size_t ws_size, hipStream_t stream) {
  const float* x       = (const float*)d_in[0];
  const float* w_theta = (const float*)d_in[1];
  const float* w_phi   = (const float*)d_in[2];
  const float* w_g     = (const float*)d_in[3];
  const float* w_o     = (const float*)d_in[4];
  const float* gamma   = (const float*)d_in[5];
  float* out = (float*)d_out;

  // Workspace carve (bf16 buffers), 256B aligned.
  char* p = (char*)d_ws;
  auto carve = [&](size_t bytes) {
    char* r = p;
    p += (bytes + 255) & ~size_t(255);
    return r;
  };
  __bf16* wcat    = (__bf16*)carve(192 * 256 * 2);
  __bf16* wo_b    = (__bf16*)carve(256 * 128 * 2);
  __bf16* thetaT  = (__bf16*)carve((size_t)16 * 4096 * 32 * 2);
  __bf16* phi_pre = (__bf16*)carve((size_t)16 * 32 * 4096 * 2);
  __bf16* g_pre   = (__bf16*)carve((size_t)16 * 128 * 4096 * 2);
  __bf16* phi_p   = (__bf16*)carve((size_t)16 * 1024 * 32 * 2);
  __bf16* g_p     = (__bf16*)carve((size_t)16 * 128 * 1024 * 2);

  k_pack_weights<<<(192 * 256 + 256 * 128 + 255) / 256, 256, 0, stream>>>(
      w_theta, w_phi, w_g, w_o, wcat, wo_b);

  k_proj<<<dim3(64, 16), 256, 64 * 256 * 2, stream>>>(x, wcat, thetaT, phi_pre,
                                                      g_pre);

  int pool_elems = 16 * 1024 * 32 + 16 * 128 * 1024;
  k_pool<<<(pool_elems + 255) / 256, 256, 0, stream>>>(phi_pre, g_pre, phi_p,
                                                       g_p);

  k_attn<<<dim3(128, 16), 256, ATTN_LDS, stream>>>(x, thetaT, phi_p, g_p, wo_b,
                                                   gamma, out);
}
